// AutoformerDecoderLayer_23716809408598
// MI455X (gfx1250) — matrix operand order repaired and checked by reading, not compiled, not run on hardware
//
#include <hip/hip_runtime.h>
#include <cstdint>

// ---------------------------------------------------------------------------
// CDNA5 / gfx1250 wave32 WMMA types
// ---------------------------------------------------------------------------
typedef __attribute__((ext_vector_type(16))) _Float16 v16h;
typedef __attribute__((ext_vector_type(8)))  float    v8f;

union FragA { v16h v; _Float16 h[16]; unsigned u[8]; };
union FragC { v8f  v; float     f[8];  };
union H4    { _Float16 h[4]; uint2 u2; };
union H8    { _Float16 h[8]; uint4 u4; };

__device__ __forceinline__ float gelu_exact(float x) {
    return 0.5f * x * (1.0f + erff(x * 0.70710678118654752f));
}

__device__ __forceinline__ uint2 cvt4_f32_f16(float4 f) {
    H4 t;
    t.h[0] = (_Float16)f.x; t.h[1] = (_Float16)f.y;
    t.h[2] = (_Float16)f.z; t.h[3] = (_Float16)f.w;
    return t.u2;
}

// ---------------------------------------------------------------------------
// GEMM:  C[M,N] = A[M,K] @ W[N,K]^T + bias[N]   (+ epilogue)
//   epilogue 0: none   1: exact GELU   2: += residual[M,N]
//   output: outH (f16) if non-null else outF (f32)
// Block: 256 threads = 8 waves; block tile 128x64; wave grid 4x2, each wave a
// 32x32 output block (4 WMMAs / 32-wide K step). Staging is software-
// pipelined: next K slab is loaded (b128) into registers while the current
// slab feeds the matrix pipes from LDS.
// ---------------------------------------------------------------------------
__global__ __launch_bounds__(256)
void gemm_wmma(const float* __restrict__ A, const float* __restrict__ W,
               const float* __restrict__ bias, const float* __restrict__ residual,
               float* __restrict__ outF, _Float16* __restrict__ outH,
               int M, int N, int K, int epilogue)
{
    __shared__ __align__(16) _Float16 Alds[128][36];  // 72 B stride: b64-aligned,
    __shared__ __align__(16) _Float16 Wlds[64][36];   // conflict-free frag reads

    const int tid    = threadIdx.x;
    const int lane   = tid & 31;
    const int wave   = tid >> 5;
    const int wm     = (wave >> 1) << 5;         // 0,32,64,96
    const int wn     = (wave & 1) << 5;          // 0,32
    const int bm     = blockIdx.x << 7;
    const int bn     = blockIdx.y << 6;
    const int ln16   = lane & 15;
    const int hiHalf = (lane >= 16);

    FragC c00, c01, c10, c11;
    {
        const float b0 = bias[bn + wn + ln16];
        const float b1 = bias[bn + wn + 16 + ln16];
#pragma unroll
        for (int r = 0; r < 8; ++r) {
            c00.f[r] = b0; c01.f[r] = b1;
            c10.f[r] = b0; c11.f[r] = b1;
        }
    }

    // staging coords: thread covers 4 consecutive cols of 6 rows (4 A + 2 W)
    const int srow = tid >> 3;            // 0..31
    const int scol = (tid & 7) << 2;      // 0,4,...,28
    const float* Abase = A + (size_t)bm * K;
    const float* Wbase = W + (size_t)bn * K;

    float4 av0, av1, av2, av3, wv0, wv1;
    auto load_slab = [&](int k0) {
        av0 = *(const float4*)(Abase + (size_t)(srow)      * K + k0 + scol);
        av1 = *(const float4*)(Abase + (size_t)(srow + 32) * K + k0 + scol);
        av2 = *(const float4*)(Abase + (size_t)(srow + 64) * K + k0 + scol);
        av3 = *(const float4*)(Abase + (size_t)(srow + 96) * K + k0 + scol);
        wv0 = *(const float4*)(Wbase + (size_t)(srow)      * K + k0 + scol);
        wv1 = *(const float4*)(Wbase + (size_t)(srow + 32) * K + k0 + scol);
    };
    auto store_slab = [&]() {
        *(uint2*)(&Alds[srow]     [scol]) = cvt4_f32_f16(av0);
        *(uint2*)(&Alds[srow + 32][scol]) = cvt4_f32_f16(av1);
        *(uint2*)(&Alds[srow + 64][scol]) = cvt4_f32_f16(av2);
        *(uint2*)(&Alds[srow + 96][scol]) = cvt4_f32_f16(av3);
        *(uint2*)(&Wlds[srow]     [scol]) = cvt4_f32_f16(wv0);
        *(uint2*)(&Wlds[srow + 32][scol]) = cvt4_f32_f16(wv1);
    };

    load_slab(0);
    for (int k0 = 0; k0 < K; k0 += 32) {
        store_slab();
        __syncthreads();
        if (k0 + 32 < K) load_slab(k0 + 32);   // overlap with WMMA below

        // A fragments: 16x32, lane = M row, halves = K (0..15 | 16..31)
        FragA a0, a1, b0f, b1f;
        {
            const int akb = hiHalf ? 16 : 0;
#pragma unroll
            for (int v = 0; v < 8; ++v) {
                const int kk = akb + (v << 1);
                a0.h[2 * v]     = Alds[wm + ln16][kk];
                a0.h[2 * v + 1] = Alds[wm + ln16][kk + 1];
                a1.h[2 * v]     = Alds[wm + 16 + ln16][kk];
                a1.h[2 * v + 1] = Alds[wm + 16 + ln16][kk + 1];
            }
        }
        // B fragments: 32x16, lane = N col, halves = K (0..15 | 16..31)
        {
            const int bkb = hiHalf ? 16 : 0;
#pragma unroll
            for (int j = 0; j < 16; ++j) {
                b0f.h[j] = Wlds[wn + ln16][bkb + j];
                b1f.h[j] = Wlds[wn + 16 + ln16][bkb + j];
            }
        }
        c00.v = __builtin_amdgcn_wmma_f32_16x16x32_f16(false, a0.v, false, b0f.v,
                                                       (short)0, c00.v, false, false);
        c01.v = __builtin_amdgcn_wmma_f32_16x16x32_f16(false, a0.v, false, b1f.v,
                                                       (short)0, c01.v, false, false);
        c10.v = __builtin_amdgcn_wmma_f32_16x16x32_f16(false, a1.v, false, b0f.v,
                                                       (short)0, c10.v, false, false);
        c11.v = __builtin_amdgcn_wmma_f32_16x16x32_f16(false, a1.v, false, b1f.v,
                                                       (short)0, c11.v, false, false);
        __syncthreads();
    }

    // epilogue + store (C layout: VGPR r -> row r (lanes 0-15) / r+8 (16-31))
    const int n0 = bn + wn + ln16;
    const int n1 = n0 + 16;
    const int mb = bm + wm + (hiHalf << 3);
#pragma unroll
    for (int r = 0; r < 8; ++r) {
        const int m0 = mb + r;
        const int m1 = m0 + 16;
        float v00 = c00.f[r], v01 = c01.f[r];
        float v10 = c10.f[r], v11 = c11.f[r];
        if (epilogue == 1) {
            v00 = gelu_exact(v00); v01 = gelu_exact(v01);
            v10 = gelu_exact(v10); v11 = gelu_exact(v11);
        } else if (epilogue == 2) {
            v00 += residual[(size_t)m0 * N + n0];
            v01 += residual[(size_t)m0 * N + n1];
            v10 += residual[(size_t)m1 * N + n0];
            v11 += residual[(size_t)m1 * N + n1];
        }
        if (outH) {
            outH[(size_t)m0 * N + n0] = (_Float16)v00;
            outH[(size_t)m0 * N + n1] = (_Float16)v01;
            outH[(size_t)m1 * N + n0] = (_Float16)v10;
            outH[(size_t)m1 * N + n1] = (_Float16)v11;
        } else {
            outF[(size_t)m0 * N + n0] = v00;
            outF[(size_t)m0 * N + n1] = v01;
            outF[(size_t)m1 * N + n0] = v10;
            outF[(size_t)m1 * N + n1] = v11;
        }
    }
}

// ---------------------------------------------------------------------------
// Flash-style attention with ALiBi-like bias, per (b, h, 64 q-rows) block.
// Q/K/V are f16, laid out (B, L, H*DK) with head h in cols [h*64, h*64+64).
// 128 threads = 4 waves; each wave owns 16 q rows. Streaming online softmax.
// K/V chunk staging is software-pipelined (b128 register loads overlap WMMA).
// ---------------------------------------------------------------------------
__global__ __launch_bounds__(128)
void attn_wmma(const _Float16* __restrict__ Q, const _Float16* __restrict__ Kp,
               const _Float16* __restrict__ V, float* __restrict__ O)
{
    const int L = 1024, D = 512, DK = 64;
    const int bx   = blockIdx.x;
    const int qt   = bx & 15;          // L/64 = 16 q-tiles
    const int h    = (bx >> 4) & 7;
    const int b    = bx >> 7;
    const int tid  = threadIdx.x;
    const int lane = tid & 31;
    const int wave = tid >> 5;
    const int ln16 = lane & 15;
    const int hiHalf = (lane >= 16);
    const int rowAdd = hiHalf ? 8 : 0;

    const int qBase = qt * 64 + wave * 16;
    const size_t headOff = (size_t)b * L * D + (size_t)h * DK;

    __shared__ __align__(16) _Float16 Klds[32][72];   // [key][dk], 144 B stride
    __shared__ __align__(16) _Float16 Vlds[64][34];   // [dk][key] transposed
    __shared__ __align__(16) _Float16 Plds[4][16][34];

    // Preload Q A-fragments for both dk chunks (dk 0-31 and 32-63), dword loads
    FragA qa0, qa1;
    {
        const unsigned* q32 =
            (const unsigned*)(Q + headOff + (size_t)(qBase + ln16) * D);
        const int akd = hiHalf ? 8 : 0;
#pragma unroll
        for (int v = 0; v < 8; ++v) {
            const int dw = akd + v;
            qa0.u[v] = q32[dw];
            qa1.u[v] = q32[16 + dw];
        }
    }

    float m_run[8], l_run[8];
    FragC o0, o1, o2, o3;
#pragma unroll
    for (int r = 0; r < 8; ++r) {
        m_run[r] = -1e30f; l_run[r] = 0.0f;
        o0.f[r] = o1.f[r] = o2.f[r] = o3.f[r] = 0.0f;
    }

    // staging coords: thread covers 8 dk of key0 and key0+16 for K and V
    const int key0 = tid >> 3;           // 0..15
    const int dk8  = (tid & 7) << 3;     // 0,8,...,56
    H8 kreg0, kreg1, vreg0, vreg1;
    auto load_kv = [&](int kc) {
        const _Float16* kb = Kp + headOff + (size_t)kc * D;
        const _Float16* vb = V  + headOff + (size_t)kc * D;
        kreg0.u4 = *(const uint4*)(kb + (size_t)key0 * D + dk8);
        kreg1.u4 = *(const uint4*)(kb + (size_t)(key0 + 16) * D + dk8);
        vreg0.u4 = *(const uint4*)(vb + (size_t)key0 * D + dk8);
        vreg1.u4 = *(const uint4*)(vb + (size_t)(key0 + 16) * D + dk8);
    };
    auto store_kv = [&]() {
        *(uint4*)(&Klds[key0][dk8])      = kreg0.u4;
        *(uint4*)(&Klds[key0 + 16][dk8]) = kreg1.u4;
#pragma unroll
        for (int i = 0; i < 8; ++i) {
            Vlds[dk8 + i][key0]      = vreg0.h[i];
            Vlds[dk8 + i][key0 + 16] = vreg1.h[i];
        }
    };

    load_kv(0);
    for (int kc = 0; kc < L; kc += 32) {
        store_kv();
        __syncthreads();
        if (kc + 32 < L) load_kv(kc + 32);   // overlap with compute below

        // scores S = Q @ K^T : two 16x16 key tiles, 2 dk-chunk WMMAs each
        FragC s0, s1;
#pragma unroll
        for (int r = 0; r < 8; ++r) { s0.f[r] = 0.0f; s1.f[r] = 0.0f; }
        {
            const int bkb = hiHalf ? 16 : 0;
            FragA kb00, kb10, kb01, kb11;
#pragma unroll
            for (int j = 0; j < 16; ++j) {
                kb00.h[j] = Klds[ln16][bkb + j];
                kb10.h[j] = Klds[16 + ln16][bkb + j];
                kb01.h[j] = Klds[ln16][32 + bkb + j];
                kb11.h[j] = Klds[16 + ln16][32 + bkb + j];
            }
            s0.v = __builtin_amdgcn_wmma_f32_16x16x32_f16(false, qa0.v, false, kb00.v,
                                                          (short)0, s0.v, false, false);
            s0.v = __builtin_amdgcn_wmma_f32_16x16x32_f16(false, qa1.v, false, kb01.v,
                                                          (short)0, s0.v, false, false);
            s1.v = __builtin_amdgcn_wmma_f32_16x16x32_f16(false, qa0.v, false, kb10.v,
                                                          (short)0, s1.v, false, false);
            s1.v = __builtin_amdgcn_wmma_f32_16x16x32_f16(false, qa1.v, false, kb11.v,
                                                          (short)0, s1.v, false, false);
        }

        // scale + ALiBi bias + online softmax (row = fixed VGPR index,
        // row values live in a 16-lane half -> width-16 xor reductions)
#pragma unroll
        for (int r = 0; r < 8; ++r) {
            const float qg  = (float)(qBase + rowAdd + r);
            const float k0g = (float)(kc + ln16);
            const float k1g = k0g + 16.0f;
            float v0 = s0.f[r] * 0.125f - 0.1f * fabsf(k0g - qg);
            float v1 = s1.f[r] * 0.125f - 0.1f * fabsf(k1g - qg);
            float mx = fmaxf(v0, v1);
#pragma unroll
            for (int off = 8; off >= 1; off >>= 1)
                mx = fmaxf(mx, __shfl_xor(mx, off, 16));
            const float mnew  = fmaxf(m_run[r], mx);
            const float alpha = __expf(m_run[r] - mnew);
            const float p0 = __expf(v0 - mnew);
            const float p1 = __expf(v1 - mnew);
            float rs = p0 + p1;
#pragma unroll
            for (int off = 8; off >= 1; off >>= 1)
                rs += __shfl_xor(rs, off, 16);
            m_run[r] = mnew;
            l_run[r] = l_run[r] * alpha + rs;
            o0.f[r] *= alpha; o1.f[r] *= alpha;
            o2.f[r] *= alpha; o3.f[r] *= alpha;
            s0.f[r] = p0; s1.f[r] = p1;
        }

        // transpose P through per-wave LDS: C-layout -> A-fragment layout
#pragma unroll
        for (int r = 0; r < 8; ++r) {
            const int row = r + rowAdd;
            Plds[wave][row][ln16]      = (_Float16)s0.f[r];
            Plds[wave][row][16 + ln16] = (_Float16)s1.f[r];
        }
        asm volatile("s_wait_dscnt 0x0" ::: "memory");  // in-wave LDS RAW fence
        FragA pa;
        {
            const int akb = hiHalf ? 16 : 0;
#pragma unroll
            for (int v = 0; v < 8; ++v) {
                const int kk = akb + (v << 1);
                pa.h[2 * v]     = Plds[wave][ln16][kk];
                pa.h[2 * v + 1] = Plds[wave][ln16][kk + 1];
            }
        }

        // O += P @ V : 4 output dk tiles, contraction = 32 keys (one WMMA each)
        {
            const int bkb = hiHalf ? 16 : 0;
#pragma unroll
            for (int t = 0; t < 4; ++t) {
                FragA vb;
#pragma unroll
                for (int j = 0; j < 16; ++j)
                    vb.h[j] = Vlds[t * 16 + ln16][bkb + j];
                FragC* ot = (t == 0) ? &o0 : (t == 1) ? &o1 : (t == 2) ? &o2 : &o3;
                ot->v = __builtin_amdgcn_wmma_f32_16x16x32_f16(false, pa.v, false, vb.v,
                                                               (short)0, ot->v, false, false);
            }
        }
        __syncthreads();
    }

    // normalize and store f32 attention output
#pragma unroll
    for (int r = 0; r < 8; ++r) {
        const float inv = 1.0f / l_run[r];
        const int m = qBase + rowAdd + r;
        const size_t base = headOff + (size_t)m * D;
        O[base + 0  + ln16] = o0.f[r] * inv;
        O[base + 16 + ln16] = o1.f[r] * inv;
        O[base + 32 + ln16] = o2.f[r] * inv;
        O[base + 48 + ln16] = o3.f[r] * inv;
    }
}

// ---------------------------------------------------------------------------
// Fused: 25-tap zero-padded moving average over L, then LayerNorm over D.
// One block per (b, l); 256 threads, 2 channels each (D = 512).
// ---------------------------------------------------------------------------
__global__ __launch_bounds__(256)
void movavg_ln(const float* __restrict__ in, const float* __restrict__ gamma,
               const float* __restrict__ beta, float* __restrict__ out)
{
    const int L = 1024, D = 512;
    const int b   = blockIdx.x >> 10;
    const int l   = blockIdx.x & 1023;
    const int tid = threadIdx.x;
    const float* base = in + (size_t)b * L * D;

    const int d0 = tid, d1 = tid + 256;
    const int jlo = (l - 12 < 0) ? 0 : l - 12;
    const int jhi = (l + 12 > L - 1) ? L - 1 : l + 12;
    float a0 = 0.0f, a1 = 0.0f;
    for (int j = jlo; j <= jhi; ++j) {
        a0 += base[(size_t)j * D + d0];
        a1 += base[(size_t)j * D + d1];
    }
    a0 *= (1.0f / 25.0f);     // count_include_pad: always divide by 25
    a1 *= (1.0f / 25.0f);

    __shared__ float red[256];
    red[tid] = a0 + a1;
    __syncthreads();
    for (int st = 128; st > 0; st >>= 1) {
        if (tid < st) red[tid] += red[tid + st];
        __syncthreads();
    }
    const float mu = red[0] * (1.0f / 512.0f);
    __syncthreads();

    const float c0 = a0 - mu, c1 = a1 - mu;
    red[tid] = c0 * c0 + c1 * c1;
    __syncthreads();
    for (int st = 128; st > 0; st >>= 1) {
        if (tid < st) red[tid] += red[tid + st];
        __syncthreads();
    }
    const float var = red[0] * (1.0f / 512.0f);
    const float inv = rsqrtf(var + 1e-5f);

    float* obase = out + (size_t)b * L * D + (size_t)l * D;
    obase[d0] = c0 * inv * gamma[d0] + beta[d0];
    obase[d1] = c1 * inv * gamma[d1] + beta[d1];
}

// ---------------------------------------------------------------------------
// Host-side orchestration
// ---------------------------------------------------------------------------
extern "C" void kernel_launch(void* const* d_in, const int* in_sizes, int n_in,
                              void* d_out, int out_size, void* d_ws, size_t ws_size,
                              hipStream_t stream)
{
    (void)in_sizes; (void)n_in; (void)out_size; (void)ws_size;
    const int B = 8, L = 1024, D = 512, DFF = 2048;
    const int M = B * L;   // 8192

    const float* x     = (const float*)d_in[0];
    const float* enc   = (const float*)d_in[1];
    const float* sa_Wq = (const float*)d_in[2];
    const float* sa_bq = (const float*)d_in[3];
    const float* sa_Wk = (const float*)d_in[4];
    const float* sa_bk = (const float*)d_in[5];
    const float* sa_Wv = (const float*)d_in[6];
    const float* sa_bv = (const float*)d_in[7];
    const float* sa_Wo = (const float*)d_in[8];
    const float* sa_bo = (const float*)d_in[9];
    const float* ca_Wq = (const float*)d_in[10];
    const float* ca_bq = (const float*)d_in[11];
    const float* ca_Wk = (const float*)d_in[12];
    const float* ca_bk = (const float*)d_in[13];
    const float* ca_Wv = (const float*)d_in[14];
    const float* ca_bv = (const float*)d_in[15];
    const float* ca_Wo = (const float*)d_in[16];
    const float* ca_bo = (const float*)d_in[17];
    const float* ff_W1 = (const float*)d_in[18];
    const float* ff_b1 = (const float*)d_in[19];
    const float* ff_W2 = (const float*)d_in[20];
    const float* ff_b2 = (const float*)d_in[21];
    const float* n1_g  = (const float*)d_in[22];
    const float* n1_b  = (const float*)d_in[23];
    const float* n2_g  = (const float*)d_in[24];
    const float* n2_b  = (const float*)d_in[25];
    const float* n3_g  = (const float*)d_in[26];
    const float* n3_b  = (const float*)d_in[27];

    char* ws = (char*)d_ws;
    const size_t MB = 1024ull * 1024ull;
    _Float16* Q16   = (_Float16*)(ws);             // 8 MiB (8192*512*2)
    _Float16* K16   = (_Float16*)(ws + 8  * MB);   // 8 MiB
    _Float16* V16   = (_Float16*)(ws + 16 * MB);   // 8 MiB
    float*    attnO = (float*)   (ws + 24 * MB);   // 16 MiB
    float*    ybuf  = (float*)   (ws + 40 * MB);   // 16 MiB
    float*    xcur  = (float*)   (ws + 56 * MB);   // 16 MiB
    float*    hbuf  = (float*)   (ws + 72 * MB);   // 64 MiB (8192*2048*4)

    dim3 gD  (M / 128, D   / 64);       // (64, 8)
    dim3 gDFF(M / 128, DFF / 64);       // (64, 32)
    dim3 aGrid(B * 8 * (L / 64));       // 1024 blocks
    dim3 mGrid(B * L);                  // 8192 blocks

    // ---- self attention ----
    gemm_wmma<<<gD, 256, 0, stream>>>(x, sa_Wq, sa_bq, nullptr, nullptr, Q16, M, D, D, 0);
    gemm_wmma<<<gD, 256, 0, stream>>>(x, sa_Wk, sa_bk, nullptr, nullptr, K16, M, D, D, 0);
    gemm_wmma<<<gD, 256, 0, stream>>>(x, sa_Wv, sa_bv, nullptr, nullptr, V16, M, D, D, 0);
    attn_wmma<<<aGrid, 128, 0, stream>>>(Q16, K16, V16, attnO);
    gemm_wmma<<<gD, 256, 0, stream>>>(attnO, sa_Wo, sa_bo, x, ybuf, nullptr, M, D, D, 2);
    movavg_ln<<<mGrid, 256, 0, stream>>>(ybuf, n1_g, n1_b, xcur);

    // ---- cross attention ----
    gemm_wmma<<<gD, 256, 0, stream>>>(xcur, ca_Wq, ca_bq, nullptr, nullptr, Q16, M, D, D, 0);
    gemm_wmma<<<gD, 256, 0, stream>>>(enc,  ca_Wk, ca_bk, nullptr, nullptr, K16, M, D, D, 0);
    gemm_wmma<<<gD, 256, 0, stream>>>(enc,  ca_Wv, ca_bv, nullptr, nullptr, V16, M, D, D, 0);
    attn_wmma<<<aGrid, 128, 0, stream>>>(Q16, K16, V16, attnO);
    gemm_wmma<<<gD, 256, 0, stream>>>(attnO, ca_Wo, ca_bo, xcur, ybuf, nullptr, M, D, D, 2);
    movavg_ln<<<mGrid, 256, 0, stream>>>(ybuf, n2_g, n2_b, xcur);

    // ---- feed forward ----
    gemm_wmma<<<gDFF, 256, 0, stream>>>(xcur, ff_W1, ff_b1, nullptr, hbuf, nullptr, M, DFF, D, 1);
    gemm_wmma<<<gD,   256, 0, stream>>>(hbuf, ff_W2, ff_b2, xcur, ybuf, nullptr, M, D, DFF, 2);
    movavg_ln<<<mGrid, 256, 0, stream>>>(ybuf, n3_g, n3_b, (float*)d_out);
}
